// SChunkTransformerEncoderLayer_48567490183179
// MI455X (gfx1250) — matrix operand order repaired and checked by reading, not compiled, not run on hardware
//
#include <hip/hip_runtime.h>

// ---------------------------------------------------------------------------
// SChunkTransformerEncoderLayer for gfx1250 (MI455X), bf16 WMMA pipeline.
// B=16, L=2048, C=512, HEADS=4, DK=128, CHUNK=16, SHIFT=8, nW=128.
// Memory-bound layer -> bf16 activations between GEMMs, fp32 accumulation,
// fused QKV GEMM (one pass over activations), async-to-LDS tile staging.
// ---------------------------------------------------------------------------

#define L_SEQ   2048
#define CDIM    512
#define BATCH   16
#define HEADSN  4
#define DKH     128
#define NWIN    128
#define CHUNKW  16
#define SHIFTW  8
#define MROWS   (BATCH * L_SEQ)       // 32768 token rows
#define FFDIM   (4 * CDIM)            // 2048
#define QKVN    (3 * CDIM)            // 1536 fused QKV width

typedef __attribute__((ext_vector_type(16))) __bf16 v16bf;
typedef __attribute__((ext_vector_type(8)))  float  v8f;

__device__ __forceinline__ v16bf frag16(uint4 lo, uint4 hi) {
    union { uint4 u[2]; v16bf v; } x;
    x.u[0] = lo; x.u[1] = hi;
    return x.v;
}

// fp32 -> bf16 (round-to-nearest-even) as raw ushort.
__device__ __forceinline__ unsigned short f2b(float f) {
    unsigned int u = __float_as_uint(f);
    u += 0x7fffu + ((u >> 16) & 1u);
    return (unsigned short)(u >> 16);
}

__device__ __forceinline__ v8f wmma_bf16(v16bf a, v16bf b, v8f c) {
    return __builtin_amdgcn_wmma_f32_16x16x32_bf16(
        false, a, false, b, (short)0, c, false, false);
}

// Per-lane async global->LDS copy of 16B (CDNA5, tracked by ASYNCcnt).
// LDS offset = low 32 bits of the flat shared-mem address.
__device__ __forceinline__ void async_b128(const void* gptr, void* lptr) {
    unsigned long long ga = (unsigned long long)(size_t)gptr;
    unsigned           la = (unsigned)(size_t)lptr;
    asm volatile("global_load_async_to_lds_b128 %0, %1, off"
                 :: "v"(la), "v"(ga) : "memory");
}
__device__ __forceinline__ void wait_async0() {
    asm volatile("s_wait_asynccnt 0" ::: "memory");
}

// ---------------------------------------------------------------------------
// fp32 -> bf16 bulk convert (plain, and column-packed for fused QKV weight)
// ---------------------------------------------------------------------------
__global__ __launch_bounds__(256) void f2bf_kernel(const float* __restrict__ in,
                                                   unsigned short* __restrict__ out,
                                                   int n) {
    int i = blockIdx.x * 256 + threadIdx.x;
    if (i < n) out[i] = f2b(in[i]);
}

__global__ __launch_bounds__(256)
void f2bf_pack_kernel(const float* __restrict__ in, unsigned short* __restrict__ out,
                      int n, int ncol, int ldout, int coloff) {
    int i = blockIdx.x * 256 + threadIdx.x;
    if (i < n) {
        int r = i / ncol, c = i - r * ncol;
        out[(size_t)r * ldout + coloff + c] = f2b(in[i]);
    }
}

__global__ __launch_bounds__(256)
void pack_bias3_kernel(const float* __restrict__ a, const float* __restrict__ b,
                       const float* __restrict__ c, float* __restrict__ out) {
    int i = blockIdx.x * 256 + threadIdx.x;
    if (i < CDIM)            out[i] = a[i];
    else if (i < 2 * CDIM)   out[i] = b[i - CDIM];
    else if (i < 3 * CDIM)   out[i] = c[i - 2 * CDIM];
}

// ---------------------------------------------------------------------------
// LayerNorm + cyclic shift. One wave per 512-elem row.
// ---------------------------------------------------------------------------
__global__ __launch_bounds__(256)
void ln_shift_kernel(const float* __restrict__ x, const float* __restrict__ g,
                     const float* __restrict__ be, unsigned short* __restrict__ out,
                     const float* __restrict__ mask_in, float* __restrict__ smask_out,
                     int shift) {
    int wave = threadIdx.x >> 5;
    int lane = threadIdx.x & 31;
    int row  = blockIdx.x * 8 + wave;
    int b = row / L_SEQ;
    int l = row - b * L_SEQ;
    int ldst = (l - shift + L_SEQ) & (L_SEQ - 1);
    size_t src = (size_t)row * CDIM;
    size_t dst = ((size_t)b * L_SEQ + ldst) * CDIM;

    float vals[16];
    float s = 0.f, s2 = 0.f;
#pragma unroll
    for (int j = 0; j < 16; ++j) {
        float v = x[src + lane + 32 * j];
        vals[j] = v; s += v; s2 += v * v;
    }
#pragma unroll
    for (int o = 1; o < 32; o <<= 1) {
        s  += __shfl_xor(s,  o, 32);
        s2 += __shfl_xor(s2, o, 32);
    }
    float mean = s * (1.f / CDIM);
    float var  = s2 * (1.f / CDIM) - mean * mean;
    float inv  = rsqrtf(var + 1e-12f);
#pragma unroll
    for (int j = 0; j < 16; ++j) {
        int c = lane + 32 * j;
        float v = (vals[j] - mean) * inv * g[c] + be[c];
        out[dst + c] = f2b(v);
    }
    if (smask_out != nullptr && lane == 0)
        smask_out[(size_t)b * L_SEQ + ldst] = mask_in[(size_t)b * L_SEQ + l];
}

// ---------------------------------------------------------------------------
// bf16 WMMA GEMM: C[M,N] = A[M,K] * B[K,N] + bias, epilogues:
//   EPI 0: out_bf16 = (relu?) result            (QKV / FFN1)
//   EPI 1: x1[b,(l+S)%L] = x + res*mask         (Wo, reverse shift + residual)
//   EPI 2: d_out = resid + result               (FFN2, final residual)
// Block tile 128x64, BK=32, 8 waves (4x2), 2x2 WMMA tiles per wave.
// A tile staged via global_load_async_to_lds_b128 (ASYNCcnt); B tile is
// transposed into LDS so both fragments load as contiguous 16B vectors.
// ---------------------------------------------------------------------------
#define BM 128
#define BN 64
#define BK 32
#define LDSA 40
#define LDSB 40

template <int EPI>
__global__ __launch_bounds__(256)
void gemm_bf16_kernel(const unsigned short* __restrict__ A,
                      const unsigned short* __restrict__ Bm,
                      const float* __restrict__ bias,
                      int M, int N, int K,
                      unsigned short* __restrict__ outb,
                      float* __restrict__ outf,
                      const float* __restrict__ resid,
                      const float* __restrict__ maskv,
                      int relu, int shift) {
    __shared__ __align__(16) unsigned short As[BM * LDSA];
    __shared__ __align__(16) unsigned short Bs[BN * LDSB];

    int tid  = threadIdx.x;
    int wave = tid >> 5;
    int lane = tid & 31;
    int wm = wave >> 1, wn = wave & 1;
    int bm0 = blockIdx.x * BM;
    int bn0 = blockIdx.y * BN;
    int half = lane >> 4;
    int l16  = lane & 15;
    int koffA = half * 16;    // A frag: K {koffA..+15} contiguous
    int koffB = half * 16;    // B frag: K {koffB..+15} contiguous

    v8f acc[2][2];
#pragma unroll
    for (int i = 0; i < 2; ++i)
#pragma unroll
        for (int j = 0; j < 2; ++j) acc[i][j] = (v8f){0.f,0.f,0.f,0.f,0.f,0.f,0.f,0.f};

    int arow = tid >> 1;
    int acol = (tid & 1) << 4;

    for (int k0 = 0; k0 < K; k0 += BK) {
        // ---- stage A tile (128x32): 2x async 16B copies per thread ----
        {
            const unsigned short* gp = A + (size_t)(bm0 + arow) * K + k0 + acol;
            async_b128(gp,     &As[arow * LDSA + acol]);
            async_b128(gp + 8, &As[arow * LDSA + acol + 8]);
        }
        // ---- stage B tile (32x64), transposed into LDS as [n][k] ----
        {
            int kk = tid >> 3;
            int n0 = (tid & 7) << 3;
            const unsigned short* gp = Bm + (size_t)(k0 + kk) * N + bn0 + n0;
            uint4 d = *(const uint4*)gp;
            if (k0 + BK < K) __builtin_prefetch(gp + (size_t)BK * N, 0, 3);
            const unsigned short* e = (const unsigned short*)&d;
#pragma unroll
            for (int j = 0; j < 8; ++j) Bs[(n0 + j) * LDSB + kk] = e[j];
        }
        wait_async0();
        __syncthreads();
#pragma unroll
        for (int tm = 0; tm < 2; ++tm) {
            int m = wm * 32 + tm * 16 + l16;
            v16bf a = frag16(*(const uint4*)&As[m * LDSA + koffA],
                             *(const uint4*)&As[m * LDSA + koffA + 8]);
#pragma unroll
            for (int tn = 0; tn < 2; ++tn) {
                int n = wn * 32 + tn * 16 + l16;
                v16bf bfr = frag16(*(const uint4*)&Bs[n * LDSB + koffB],
                                   *(const uint4*)&Bs[n * LDSB + koffB + 8]);
                acc[tm][tn] = wmma_bf16(a, bfr, acc[tm][tn]);
            }
        }
        __syncthreads();
    }

    // ---- epilogue: C layout: vgpr r -> M = r + 8*half, N = l16 ----
#pragma unroll
    for (int tm = 0; tm < 2; ++tm)
#pragma unroll
        for (int tn = 0; tn < 2; ++tn)
#pragma unroll
            for (int r = 0; r < 8; ++r) {
                int gm = bm0 + wm * 32 + tm * 16 + r + half * 8;
                int gn = bn0 + wn * 32 + tn * 16 + l16;
                float v = acc[tm][tn][r] + bias[gn];
                if (EPI == 0) {
                    if (relu && v < 0.f) v = 0.f;
                    outb[(size_t)gm * N + gn] = f2b(v);
                } else if (EPI == 1) {
                    int b  = gm / L_SEQ;
                    int ls = gm - b * L_SEQ;
                    int ld = (ls + shift) & (L_SEQ - 1);
                    size_t dr = (size_t)b * L_SEQ + ld;
                    float mv = maskv[dr];
                    outf[dr * N + gn] = resid[dr * N + gn] + v * mv;
                } else {
                    outf[(size_t)gm * N + gn] = resid[(size_t)gm * N + gn] + v;
                }
            }
}

// ---------------------------------------------------------------------------
// Windowed attention: one wave per (batch, window, head), reading from the
// fused QKV buffer [row][1536] (q:0, k:512, v:1024).
// ---------------------------------------------------------------------------
__global__ __launch_bounds__(32)
void attn_kernel(const unsigned short* __restrict__ qkv,
                 const float* __restrict__ smask,
                 unsigned short* __restrict__ attn_out) {
    __shared__ __align__(16) unsigned short Qs[16 * 136];
    __shared__ __align__(16) unsigned short Ks[16 * 136];
    __shared__ __align__(16) unsigned short Vt[128 * 40];   // [d][key], keys 16..31 zero
    __shared__ __align__(16) unsigned short Ps[16 * 40];    // probs, K 16..31 zero

    int idx = blockIdx.x;
    int h = idx & 3;
    int w = (idx >> 2) & (NWIN - 1);
    int b = idx >> 9;
    int lane = threadIdx.x;
    int half = lane >> 4, l16 = lane & 15;
    int koffA = half * 16, koffB = half * 16;
    size_t rowbase = (size_t)b * L_SEQ + (size_t)w * CHUNKW;
    int qoff = h * DKH;
    int koff = CDIM + h * DKH;
    int voff = 2 * CDIM + h * DKH;

    // load Q,K tiles (16x128 bf16 each)
#pragma unroll
    for (int j = 0; j < 8; ++j) {
        int f = lane + 32 * j;
        int row = f >> 4, cv = (f & 15) << 3;
        size_t g = (rowbase + row) * QKVN + cv;
        *(uint4*)&Qs[row * 136 + cv] = *(const uint4*)&qkv[g + qoff];
        *(uint4*)&Ks[row * 136 + cv] = *(const uint4*)&qkv[g + koff];
    }
    // zero Vt / Ps pad columns (keys 16..31)
    uint4 z = make_uint4(0u, 0u, 0u, 0u);
#pragma unroll
    for (int j = 0; j < 8; ++j) {
        int f = lane + 32 * j;
        int d = f >> 1;
        int c = 16 + ((f & 1) << 3);
        *(uint4*)&Vt[d * 40 + c] = z;
    }
    *(uint4*)&Ps[(lane >> 1) * 40 + 16 + ((lane & 1) << 3)] = z;
    // load V transposed: Vt[d][key]
#pragma unroll
    for (int j = 0; j < 8; ++j) {
        int f = lane + 32 * j;
        int key = f >> 4, dv = (f & 15) << 3;
        uint4 dd = *(const uint4*)&qkv[(rowbase + key) * QKVN + voff + dv];
        const unsigned short* e = (const unsigned short*)&dd;
#pragma unroll
        for (int t = 0; t < 8; ++t) Vt[(dv + t) * 40 + key] = e[t];
    }
    __syncthreads();

    // scores: Q @ K^T, K-dim 128 -> 4 WMMA steps
    v8f S = (v8f){0.f,0.f,0.f,0.f,0.f,0.f,0.f,0.f};
#pragma unroll
    for (int d0 = 0; d0 < DKH; d0 += 32) {
        v16bf a  = frag16(*(const uint4*)&Qs[l16 * 136 + d0 + koffA],
                          *(const uint4*)&Qs[l16 * 136 + d0 + koffA + 8]);
        v16bf bb = frag16(*(const uint4*)&Ks[l16 * 136 + d0 + koffB],
                          *(const uint4*)&Ks[l16 * 136 + d0 + koffB + 8]);
        S = wmma_bf16(a, bb, S);
    }

    // mask + softmax (row qi in vgpr r, spread over one 16-lane half)
    const float scale = 0.088388347648318447f;  // 1/sqrt(128)
    bool lastw = (w == NWIN - 1);
    float kmv = smask[rowbase + l16];
    bool kvalid = kmv > 0.f;
#pragma unroll
    for (int r = 0; r < 8; ++r) {
        int qi = r + half * 8;
        int kj = l16;
        float s = S[r] * scale;
        if (!lastw) { if (qi < CHUNKW / 2 - 1 && kj >= CHUNKW / 2) s -= 100.f; }
        else        { if (qi >= CHUNKW / 2 && kj >= CHUNKW / 2)    s -= 100.f; }
        if (!kvalid) s = -1e30f;
        float mx = s;
        mx = fmaxf(mx, __shfl_xor(mx, 1, 16));
        mx = fmaxf(mx, __shfl_xor(mx, 2, 16));
        mx = fmaxf(mx, __shfl_xor(mx, 4, 16));
        mx = fmaxf(mx, __shfl_xor(mx, 8, 16));
        float e = kvalid ? __expf(s - mx) : 0.f;
        float sm = e;
        sm += __shfl_xor(sm, 1, 16);
        sm += __shfl_xor(sm, 2, 16);
        sm += __shfl_xor(sm, 4, 16);
        sm += __shfl_xor(sm, 8, 16);
        Ps[qi * 40 + kj] = f2b(e / (sm + 1e-30f));
    }
    __syncthreads();

    // out = P(16x32, padded) @ V(32x128, padded): 8 n-tiles
    v16bf pa = frag16(*(const uint4*)&Ps[l16 * 40 + koffA],
                      *(const uint4*)&Ps[l16 * 40 + koffA + 8]);
#pragma unroll
    for (int nt = 0; nt < 8; ++nt) {
        v16bf bb = frag16(*(const uint4*)&Vt[(nt * 16 + l16) * 40 + koffB],
                          *(const uint4*)&Vt[(nt * 16 + l16) * 40 + koffB + 8]);
        v8f o = (v8f){0.f,0.f,0.f,0.f,0.f,0.f,0.f,0.f};
        o = wmma_bf16(pa, bb, o);
#pragma unroll
        for (int r = 0; r < 8; ++r) {
            int m = r + half * 8;
            int n = nt * 16 + l16;
            attn_out[(rowbase + m) * CDIM + h * DKH + n] = f2b(o[r]);
        }
    }
}

// ---------------------------------------------------------------------------
// Orchestration. Workspace (~232 MB) with buffer reuse.
// ---------------------------------------------------------------------------
extern "C" void kernel_launch(void* const* d_in, const int* in_sizes, int n_in,
                              void* d_out, int out_size, void* d_ws, size_t ws_size,
                              hipStream_t stream) {
    (void)in_sizes; (void)n_in; (void)out_size; (void)ws_size;

    const float* x        = (const float*)d_in[0];
    const float* mask_pad = (const float*)d_in[3];
    const float* Wq = (const float*)d_in[4];  const float* bq = (const float*)d_in[5];
    const float* Wk = (const float*)d_in[6];  const float* bk = (const float*)d_in[7];
    const float* Wv = (const float*)d_in[8];  const float* bv = (const float*)d_in[9];
    const float* Wo = (const float*)d_in[10]; const float* bo = (const float*)d_in[11];
    const float* g1 = (const float*)d_in[12]; const float* be1 = (const float*)d_in[13];
    const float* g2 = (const float*)d_in[14]; const float* be2 = (const float*)d_in[15];
    const float* W1 = (const float*)d_in[16]; const float* bf1 = (const float*)d_in[17];
    const float* W2 = (const float*)d_in[18]; const float* bf2 = (const float*)d_in[19];

    char* ws = (char*)d_ws;
    const size_t SZ_WQKV = (size_t)CDIM * QKVN * 2;            // 1.5 MB
    const size_t SZ_W    = (size_t)CDIM * CDIM * 2;            // 512 KB
    const size_t SZ_W1   = (size_t)CDIM * FFDIM * 2;           // 2 MB
    const size_t SZ_ACT  = (size_t)MROWS * CDIM * 2;           // 32 MB
    const size_t OFF_WQKV = 0;
    const size_t OFF_WO  = OFF_WQKV + SZ_WQKV;
    const size_t OFF_W1  = OFF_WO + SZ_W;
    const size_t OFF_W2  = OFF_W1 + SZ_W1;
    const size_t OFF_SM  = OFF_W2 + SZ_W1;                     // 128 KB
    const size_t OFF_B3  = OFF_SM + (size_t)MROWS * 4;         // 6 KB
    const size_t OFF_XS  = 8ull << 20;                         // 8 MB
    const size_t OFF_QKV = OFF_XS + SZ_ACT;                    // 96 MB fused QKV
    const size_t OFF_AT  = OFF_XS;                             // reuse xs
    const size_t OFF_X1  = OFF_QKV + (size_t)MROWS * QKVN * 2; // fp32, 64 MB
    const size_t OFF_XN2 = OFF_X1 + (size_t)MROWS * CDIM * 4;  // bf16, 32 MB
    const size_t OFF_H   = OFF_XS;                             // 128 MB reuse xs..qkv

    unsigned short* wqkv_b = (unsigned short*)(ws + OFF_WQKV);
    unsigned short* wo_b   = (unsigned short*)(ws + OFF_WO);
    unsigned short* w1_b   = (unsigned short*)(ws + OFF_W1);
    unsigned short* w2_b   = (unsigned short*)(ws + OFF_W2);
    float*          smsk   = (float*)(ws + OFF_SM);
    float*          b3     = (float*)(ws + OFF_B3);
    unsigned short* xs     = (unsigned short*)(ws + OFF_XS);
    unsigned short* qkvb   = (unsigned short*)(ws + OFF_QKV);
    unsigned short* atb    = (unsigned short*)(ws + OFF_AT);
    float*          x1     = (float*)(ws + OFF_X1);
    unsigned short* xn2    = (unsigned short*)(ws + OFF_XN2);
    unsigned short* hb     = (unsigned short*)(ws + OFF_H);

    // 1) weights -> bf16 (QKV packed into one 512x1536 matrix)
    int nw = CDIM * CDIM;
    int gw = (nw + 255) / 256;
    f2bf_pack_kernel<<<gw, 256, 0, stream>>>(Wq, wqkv_b, nw, CDIM, QKVN, 0);
    f2bf_pack_kernel<<<gw, 256, 0, stream>>>(Wk, wqkv_b, nw, CDIM, QKVN, CDIM);
    f2bf_pack_kernel<<<gw, 256, 0, stream>>>(Wv, wqkv_b, nw, CDIM, QKVN, 2 * CDIM);
    f2bf_kernel<<<gw, 256, 0, stream>>>(Wo, wo_b, nw);
    int nw1 = CDIM * FFDIM;
    f2bf_kernel<<<(nw1 + 255) / 256, 256, 0, stream>>>(W1, w1_b, nw1);
    f2bf_kernel<<<(nw1 + 255) / 256, 256, 0, stream>>>(W2, w2_b, nw1);
    pack_bias3_kernel<<<6, 256, 0, stream>>>(bq, bk, bv, b3);

    // 2) LN1 + shift(-8) -> xs (bf16), shifted key mask -> smsk
    ln_shift_kernel<<<MROWS / 8, 256, 0, stream>>>(x, g1, be1, xs, mask_pad, smsk, SHIFTW);

    // 3) fused QKV projection: one pass over xs
    dim3 gQKV(MROWS / BM, QKVN / BN);
    gemm_bf16_kernel<0><<<gQKV, 256, 0, stream>>>(xs, wqkv_b, b3, MROWS, QKVN, CDIM,
                                                  qkvb, nullptr, nullptr, nullptr, 0, 0);

    // 4) windowed attention (one wave per batch/window/head)
    attn_kernel<<<BATCH * NWIN * HEADSN, 32, 0, stream>>>(qkvb, smsk, atb);

    // 5) output projection + reverse shift + mask + residual -> x1 (fp32)
    dim3 gC(MROWS / BM, CDIM / BN);
    gemm_bf16_kernel<1><<<gC, 256, 0, stream>>>(atb, wo_b, bo, MROWS, CDIM, CDIM,
                                                nullptr, x1, x, mask_pad, 0, SHIFTW);

    // 6) LN2 -> xn2 (bf16)
    ln_shift_kernel<<<MROWS / 8, 256, 0, stream>>>(x1, g2, be2, xn2, nullptr, nullptr, 0);

    // 7) FFN: relu(xn2 @ W1 + bf1) -> hb (bf16)
    dim3 gF1(MROWS / BM, FFDIM / BN);
    gemm_bf16_kernel<0><<<gF1, 256, 0, stream>>>(xn2, w1_b, bf1, MROWS, FFDIM, CDIM,
                                                 hb, nullptr, nullptr, nullptr, 1, 0);

    // 8) FFN out + residual -> d_out (fp32)
    gemm_bf16_kernel<2><<<gC, 256, 0, stream>>>(hb, w2_b, bf2, MROWS, CDIM, FFDIM,
                                                nullptr, (float*)d_out, x1, nullptr, 0, 0);
}
